// GCN_11862699671726
// MI455X (gfx1250) — matrix operand order repaired and checked by reading, not compiled, not run on hardware
//
#include <hip/hip_runtime.h>

typedef __attribute__((ext_vector_type(8)))  float          v8f;
typedef __attribute__((ext_vector_type(16))) __bf16         v16bf;
typedef __attribute__((ext_vector_type(8)))  __bf16         v8bf;
typedef __attribute__((ext_vector_type(8)))  unsigned short us8;
typedef __attribute__((ext_vector_type(4)))  unsigned short us4;

#define N_NODES 100000
#define FEATS   256
#define HID     128
#define CLS     32
#define N_EDGES 1600000
#define M_TILES 6250   // 100000 / 16, exact

__device__ __forceinline__ unsigned short f32_to_bf16(float f) {
  unsigned int u = __float_as_uint(f);
  u += 0x7FFFu + ((u >> 16) & 1u);           // round-to-nearest-even
  return (unsigned short)(u >> 16);
}

// ---------- x (f32) -> bf16, 8 elements/thread ----------
__global__ __launch_bounds__(256) void cvt_x_kernel(const float* __restrict__ x,
                                                    unsigned short* __restrict__ xb) {
  int t = blockIdx.x * 256 + threadIdx.x;          // 3.2M threads exactly
  long base = (long)t * 8;
  float4 a = ((const float4*)x)[(long)t * 2 + 0];
  float4 b = ((const float4*)x)[(long)t * 2 + 1];
  us8 o;
  o[0] = f32_to_bf16(a.x); o[1] = f32_to_bf16(a.y);
  o[2] = f32_to_bf16(a.z); o[3] = f32_to_bf16(a.w);
  o[4] = f32_to_bf16(b.x); o[5] = f32_to_bf16(b.y);
  o[6] = f32_to_bf16(b.z); o[7] = f32_to_bf16(b.w);
  *(us8*)(xb + base) = o;
}

// ---------- degree via f32 atomics on dst ----------
__global__ __launch_bounds__(256) void degree_kernel(const long long* __restrict__ ei,
                                                     float* __restrict__ deg) {
  int e = blockIdx.x * 256 + threadIdx.x;
  if (e < N_EDGES) atomicAdd(&deg[(int)ei[N_EDGES + e]], 1.0f);
}

__global__ __launch_bounds__(256) void dinv_kernel(const float* __restrict__ deg,
                                                   float* __restrict__ dinv) {
  int i = blockIdx.x * 256 + threadIdx.x;
  if (i < N_NODES) {
    float d = deg[i];
    dinv[i] = (d > 0.0f) ? rsqrtf(fmaxf(d, 1e-12f)) : 0.0f;
  }
}

// ---------- GEMM1: h1[100000,128] = x_bf[100000,256] @ W1[256,128], WMMA bf16 ----------
// Whole W1^T lives in LDS (128 rows x 256 K = 64 KB). 8 waves/block, 1 M-tile/wave.
__global__ __launch_bounds__(256) void gemm1_wmma(const float* __restrict__ W1,
                                                  const unsigned short* __restrict__ xbf,
                                                  float* __restrict__ h1) {
  __shared__ unsigned short Wt[HID * FEATS];       // Wt[n][k], 64 KB
  const int tid = threadIdx.x;
  for (int idx = tid; idx < FEATS * HID; idx += 256) {
    int k = idx >> 7, n = idx & 127;               // W1 row-major read = coalesced
    Wt[n * FEATS + k] = f32_to_bf16(W1[idx]);
  }
  __syncthreads();

  const int wave = tid >> 5, lane = tid & 31;
  const int mtile = blockIdx.x * 8 + wave;         // wave-uniform
  const int n16 = lane & 15, half = lane >> 4;
  long row = (long)mtile * 16 + n16;               // A row for this lane
  if (row >= N_NODES) row = N_NODES - 1;           // clamp loads (stores guarded)
  const unsigned short* arow = xbf + row * FEATS;

  v8f acc[8] = {};
  #pragma unroll
  for (int kt = 0; kt < 8; ++kt) {                 // K = 8 x 32
    int kbase = kt * 32 + half * 8;
    v8bf alo = *(const v8bf*)(arow + kbase);       // K = kbase .. +7
    v8bf ahi = *(const v8bf*)(arow + kbase + 16);  // K = kbase+16 .. +23
    v16bf a;
    #pragma unroll
    for (int i = 0; i < 8; ++i) { a[i] = alo[i]; a[8 + i] = ahi[i]; }
    #pragma unroll
    for (int nt = 0; nt < 8; ++nt) {               // N = 8 x 16
      const unsigned short* bp = &Wt[(nt * 16 + n16) * FEATS + kt * 32 + half * 8];
      v8bf blo = *(const v8bf*)(bp);
      v8bf bhi = *(const v8bf*)(bp + 16);
      v16bf b;
      #pragma unroll
      for (int i = 0; i < 8; ++i) { b[i] = blo[i]; b[8 + i] = bhi[i]; }
      acc[nt] = __builtin_amdgcn_wmma_f32_16x16x32_bf16(
          false, a, false, b, (short)0, acc[nt], false, false);
    }
  }
  if (mtile < M_TILES) {
    #pragma unroll
    for (int nt = 0; nt < 8; ++nt)
      #pragma unroll
      for (int r = 0; r < 8; ++r) {
        long mm = (long)mtile * 16 + half * 8 + r;
        h1[mm * HID + nt * 16 + n16] = acc[nt][r];
      }
  }
}

// ---------- scatter layer 1: one wave per edge, 4 feats/lane ----------
__global__ __launch_bounds__(256) void scatter1(const long long* __restrict__ ei,
                                                const float* __restrict__ dinv,
                                                const float* __restrict__ h1,
                                                float* __restrict__ agg1) {
  long t = (long)blockIdx.x * 256 + threadIdx.x;   // 51.2M threads exactly
  int e = (int)(t >> 5), lane = (int)(t & 31);
  int src = (int)ei[e], dst = (int)ei[N_EDGES + e];
  float norm = dinv[src] * dinv[dst];
  float4 v = *(const float4*)(h1 + (long)src * HID + lane * 4);
  float* o = agg1 + (long)dst * HID + lane * 4;
  atomicAdd(o + 0, v.x * norm);
  atomicAdd(o + 1, v.y * norm);
  atomicAdd(o + 2, v.z * norm);
  atomicAdd(o + 3, v.w * norm);
}

// ---------- relu(agg1 + b1) -> bf16, 4 elements/thread ----------
__global__ __launch_bounds__(256) void relu_cvt(const float* __restrict__ agg1,
                                                const float* __restrict__ b1,
                                                unsigned short* __restrict__ hb) {
  int t = blockIdx.x * 256 + threadIdx.x;          // 3.2M threads exactly
  long base = (long)t * 4;
  int n = (int)(base & 127);
  float4 v = *(const float4*)(agg1 + base);
  v.x = fmaxf(v.x + b1[n + 0], 0.0f);
  v.y = fmaxf(v.y + b1[n + 1], 0.0f);
  v.z = fmaxf(v.z + b1[n + 2], 0.0f);
  v.w = fmaxf(v.w + b1[n + 3], 0.0f);
  us4 o;
  o[0] = f32_to_bf16(v.x); o[1] = f32_to_bf16(v.y);
  o[2] = f32_to_bf16(v.z); o[3] = f32_to_bf16(v.w);
  *(us4*)(hb + base) = o;
}

// ---------- GEMM2: h2[100000,32] = h_bf[100000,128] @ W2[128,32], WMMA bf16 ----------
__global__ __launch_bounds__(256) void gemm2_wmma(const float* __restrict__ W2,
                                                  const unsigned short* __restrict__ hbf,
                                                  float* __restrict__ h2) {
  __shared__ unsigned short Wt[CLS * HID];         // Wt[n][k], 8 KB
  const int tid = threadIdx.x;
  for (int idx = tid; idx < HID * CLS; idx += 256) {
    int k = idx >> 5, n = idx & 31;
    Wt[n * HID + k] = f32_to_bf16(W2[idx]);
  }
  __syncthreads();

  const int wave = tid >> 5, lane = tid & 31;
  const int mtile = blockIdx.x * 8 + wave;
  const int n16 = lane & 15, half = lane >> 4;
  long row = (long)mtile * 16 + n16;
  if (row >= N_NODES) row = N_NODES - 1;
  const unsigned short* arow = hbf + row * HID;

  v8f acc[2] = {};
  #pragma unroll
  for (int kt = 0; kt < 4; ++kt) {                 // K = 4 x 32
    int kbase = kt * 32 + half * 8;
    v8bf alo = *(const v8bf*)(arow + kbase);
    v8bf ahi = *(const v8bf*)(arow + kbase + 16);
    v16bf a;
    #pragma unroll
    for (int i = 0; i < 8; ++i) { a[i] = alo[i]; a[8 + i] = ahi[i]; }
    #pragma unroll
    for (int nt = 0; nt < 2; ++nt) {               // N = 2 x 16
      const unsigned short* bp = &Wt[(nt * 16 + n16) * HID + kt * 32 + half * 8];
      v8bf blo = *(const v8bf*)(bp);
      v8bf bhi = *(const v8bf*)(bp + 16);
      v16bf b;
      #pragma unroll
      for (int i = 0; i < 8; ++i) { b[i] = blo[i]; b[8 + i] = bhi[i]; }
      acc[nt] = __builtin_amdgcn_wmma_f32_16x16x32_bf16(
          false, a, false, b, (short)0, acc[nt], false, false);
    }
  }
  if (mtile < M_TILES) {
    #pragma unroll
    for (int nt = 0; nt < 2; ++nt)
      #pragma unroll
      for (int r = 0; r < 8; ++r) {
        long mm = (long)mtile * 16 + half * 8 + r;
        h2[mm * CLS + nt * 16 + n16] = acc[nt][r];
      }
  }
}

// ---------- scatter layer 2: one wave per edge, 1 feat/lane ----------
__global__ __launch_bounds__(256) void scatter2(const long long* __restrict__ ei,
                                                const float* __restrict__ dinv,
                                                const float* __restrict__ h2,
                                                float* __restrict__ agg2) {
  long t = (long)blockIdx.x * 256 + threadIdx.x;
  int e = (int)(t >> 5), lane = (int)(t & 31);
  int src = (int)ei[e], dst = (int)ei[N_EDGES + e];
  float norm = dinv[src] * dinv[dst];
  float v = h2[(long)src * CLS + lane] * norm;
  atomicAdd(&agg2[(long)dst * CLS + lane], v);
}

// ---------- log_softmax over 32 classes: one wave32 per row ----------
__global__ __launch_bounds__(256) void softmax_kernel(const float* __restrict__ agg2,
                                                      const float* __restrict__ b2,
                                                      float* __restrict__ out) {
  long t = (long)blockIdx.x * 256 + threadIdx.x;   // 3.2M threads exactly
  int row = (int)(t >> 5), lane = (int)(t & 31);
  float v = agg2[(long)row * CLS + lane] + b2[lane];
  float m = v;
  #pragma unroll
  for (int s = 16; s > 0; s >>= 1) m = fmaxf(m, __shfl_xor(m, s, 32));
  float e = expf(v - m);
  float sum = e;
  #pragma unroll
  for (int s = 16; s > 0; s >>= 1) sum += __shfl_xor(sum, s, 32);
  out[(long)row * CLS + lane] = v - m - logf(sum);
}

extern "C" void kernel_launch(void* const* d_in, const int* in_sizes, int n_in,
                              void* d_out, int out_size, void* d_ws, size_t ws_size,
                              hipStream_t stream) {
  const float*     x  = (const float*)d_in[0];
  const float*     W1 = (const float*)d_in[1];
  const float*     b1 = (const float*)d_in[2];
  const float*     W2 = (const float*)d_in[3];
  const float*     b2 = (const float*)d_in[4];
  const long long* ei = (const long long*)d_in[5];
  float*           out = (float*)d_out;

  // workspace layout (lifetime-aliased):
  //   deg[100K f32] | dinv[100K f32] | bufA 51.2MB (x_bf16 -> h_bf16)
  //   bufB 51.2MB (h1 -> h2) | bufC 51.2MB (agg1 -> agg2)
  char* ws = (char*)d_ws;
  float*          deg  = (float*)(ws);
  float*          dinv = (float*)(ws + 409600);
  unsigned short* bufA = (unsigned short*)(ws + 819200);
  float*          bufB = (float*)(ws + 819200 + 51200000L);
  float*          bufC = (float*)(ws + 819200 + 2 * 51200000L);

  hipMemsetAsync(deg,  0, (size_t)N_NODES * sizeof(float), stream);
  hipMemsetAsync(bufC, 0, (size_t)N_NODES * HID * sizeof(float), stream);   // agg1 = 0

  cvt_x_kernel  <<<12500,  256, 0, stream>>>(x, bufA);                      // x -> bf16
  degree_kernel <<<6250,   256, 0, stream>>>(ei, deg);
  dinv_kernel   <<<(N_NODES + 255) / 256, 256, 0, stream>>>(deg, dinv);
  gemm1_wmma    <<<782,    256, 0, stream>>>(W1, bufA, bufB);               // h1
  scatter1      <<<200000, 256, 0, stream>>>(ei, dinv, bufB, bufC);         // agg1
  relu_cvt      <<<12500,  256, 0, stream>>>(bufC, b1, bufA);               // h_bf16
  hipMemsetAsync(bufC, 0, (size_t)N_NODES * CLS * sizeof(float), stream);   // agg2 = 0
  gemm2_wmma    <<<782,    256, 0, stream>>>(W2, bufA, bufB);               // h2
  scatter2      <<<200000, 256, 0, stream>>>(ei, dinv, bufB, bufC);         // agg2
  softmax_kernel<<<12500,  256, 0, stream>>>(bufC, b2, out);
}
